// GAT_GCN_LSTM_Attention_71201967833779
// MI455X (gfx1250) — matrix operand order repaired and checked by reading, not compiled, not run on hardware
//
#include <hip/hip_runtime.h>
#include <hip/hip_bf16.h>

// ---------------- problem constants ----------------
#define Bsz   2048
#define NPG   100
#define NT    204800          // Bsz * NPG
#define Ecnt  819200
#define ETOT  (Ecnt + NT)     // edges + self loops
#define IND   256
#define HID   128
#define HEADS 4
#define DHD   32
#define CELLD 954
#define FEATD 768             // 4*HID + IND
#define MIDD  64
#define NEGV  (-1e9f)

typedef __attribute__((ext_vector_type(16))) _Float16 v16h;
typedef __attribute__((ext_vector_type(8)))  float    v8f;

// =======================================================================
// WMMA fragment loaders.
// A 16x32 f16 layout: lanes 0-15 row=lane, halves 0..15 -> K kb+0..15;
// lanes 16-31 same rows, K kb+16..31.
// B 32x16 f16 layout: lanes 0-15 col=lane K kb+0..15; lanes 16-31 K kb+16..31.
// =======================================================================

// ---- fast (unguarded, vectorized) loaders: require 16B-aligned rows ----
__device__ __forceinline__ v16h frag_a_vec(const float* __restrict__ Arow, int kb, int kg) {
  const float* p = Arow + kb + kg;
  float4 x0 = *(const float4*)(p);
  float4 x1 = *(const float4*)(p + 4);
  float4 y0 = *(const float4*)(p + 8);
  float4 y1 = *(const float4*)(p + 12);
  v16h a;
  a[0]=(_Float16)x0.x;  a[1]=(_Float16)x0.y;  a[2]=(_Float16)x0.z;  a[3]=(_Float16)x0.w;
  a[4]=(_Float16)x1.x;  a[5]=(_Float16)x1.y;  a[6]=(_Float16)x1.z;  a[7]=(_Float16)x1.w;
  a[8]=(_Float16)y0.x;  a[9]=(_Float16)y0.y;  a[10]=(_Float16)y0.z; a[11]=(_Float16)y0.w;
  a[12]=(_Float16)y1.x; a[13]=(_Float16)y1.y; a[14]=(_Float16)y1.z; a[15]=(_Float16)y1.w;
  return a;
}

__device__ __forceinline__ v16h frag_bt_vec(const float* __restrict__ Brow, int kb, int kgb) {
  const float* p = Brow + kb + kgb;
  float4 x0 = *(const float4*)(p);
  float4 x1 = *(const float4*)(p + 4);
  float4 x2 = *(const float4*)(p + 8);
  float4 x3 = *(const float4*)(p + 12);
  v16h b;
  b[0]=(_Float16)x0.x;  b[1]=(_Float16)x0.y;  b[2]=(_Float16)x0.z;  b[3]=(_Float16)x0.w;
  b[4]=(_Float16)x1.x;  b[5]=(_Float16)x1.y;  b[6]=(_Float16)x1.z;  b[7]=(_Float16)x1.w;
  b[8]=(_Float16)x2.x;  b[9]=(_Float16)x2.y;  b[10]=(_Float16)x2.z; b[11]=(_Float16)x2.w;
  b[12]=(_Float16)x3.x; b[13]=(_Float16)x3.y; b[14]=(_Float16)x3.z; b[15]=(_Float16)x3.w;
  return b;
}

// non-transposed B: column strided by ldb, coalesced across lanes
__device__ __forceinline__ v16h frag_bn_vec(const float* __restrict__ Bm, int ldb, int col,
                                            int kb, int kgb) {
  v16h b;
  const float* p = Bm + (size_t)(kb + kgb) * ldb + col;
  #pragma unroll
  for (int h = 0; h < 16; ++h) { b[h] = (_Float16)(*p); p += ldb; }
  return b;
}

// ---- guarded loaders (odd shapes only): BRANCH-FREE via clamped address +
// value select; loads execute unconditionally (clamped index stays in-bounds).
__device__ __forceinline__ v16h frag_a_g(const float* __restrict__ A, int lda, int row,
                                         int kb, int K, int lane) {
  v16h a;
  int kg = (lane >> 4) << 4;
  const float* Arow = A + (size_t)row * lda;
  #pragma unroll
  for (int h = 0; h < 8; ++h) {
    int k0 = kb + kg + h;
    int k1 = kb + 8 + kg + h;
    int k0c = (k0 < K) ? k0 : 0;
    int k1c = (k1 < K) ? k1 : 0;
    float f0 = Arow[k0c];
    float f1 = Arow[k1c];
    f0 = (k0 < K) ? f0 : 0.f;
    f1 = (k1 < K) ? f1 : 0.f;
    a[h]     = (_Float16)f0;
    a[h + 8] = (_Float16)f1;
  }
  return a;
}

__device__ __forceinline__ v16h frag_b_g(const float* __restrict__ Bm, int ldb, int col,
                                         int Nc, int kb, int K, int lane, int transB) {
  v16h b;
  int kbase = kb + ((lane >> 4) << 4);
  int colc = (col < Nc) ? col : 0;
  bool cok = col < Nc;
  #pragma unroll
  for (int h = 0; h < 16; ++h) {
    int k = kbase + h;
    int kc = (k < K) ? k : 0;
    float f = transB ? Bm[(size_t)colc * ldb + kc] : Bm[(size_t)kc * ldb + colc];
    f = (cok && k < K) ? f : 0.f;
    b[h] = (_Float16)f;
  }
  return b;
}

__device__ __forceinline__ void store_tile(v8f acc, float* __restrict__ C, int ldc, int m, int n,
                                           const float* __restrict__ bias, int beta, int relu) {
  float bv = bias ? bias[n] : 0.f;
  #pragma unroll
  for (int i = 0; i < 8; ++i) {
    float v = acc[i] + bv;
    if (beta) v += C[(size_t)(m + i) * ldc + n];
    if (relu && v < 0.f) v = 0.f;
    C[(size_t)(m + i) * ldc + n] = v;
  }
}

// =======================================================================
// FAST GEMM: C = act(A@B(^T) [+bias] [+C]); requires M%32==0, N%32==0,
// K%32==0, lda%4==0, (ldb%4==0 if TRANSB), 16B-aligned base pointers.
// block = 256 (8 waves); wave computes a 32x32 tile; block covers 64x128.
// =======================================================================
template <int TRANSB>
__global__ void gemm_wmma_fast(const float* __restrict__ A, int lda,
                               const float* __restrict__ Bm, int ldb,
                               const float* __restrict__ bias,
                               float* __restrict__ C, int ldc,
                               int M, int N, int K, int beta, int relu) {
  int lane = threadIdx.x & 31;
  int wave = threadIdx.x >> 5;
  int m0 = blockIdx.y * 64 + (wave >> 2) * 32;
  int n0 = blockIdx.x * 128 + (wave & 3) * 32;
  if (m0 >= M || n0 >= N) return;          // wave-uniform: EXEC all-ones at WMMA
  int kg  = (lane >> 4) << 4;
  int kgb = (lane >> 4) << 4;
  const float* Arow0 = A + (size_t)(m0 + (lane & 15)) * lda;
  const float* Arow1 = Arow0 + (size_t)16 * lda;
  const float* Bp0 = nullptr; const float* Bp1 = nullptr;
  if (TRANSB) {
    Bp0 = Bm + (size_t)(n0 + (lane & 15)) * ldb;
    Bp1 = Bp0 + (size_t)16 * ldb;
  }
  v8f a00 = {}, a01 = {}, a10 = {}, a11 = {};
  for (int kb = 0; kb < K; kb += 32) {
    if (kb + 32 < K) {                      // prefetch next K-chunk (global_prefetch_b8)
      __builtin_prefetch(Arow0 + kb + 32 + kg, 0, 0);
      __builtin_prefetch(Arow1 + kb + 32 + kg, 0, 0);
      if (TRANSB) {
        __builtin_prefetch(Bp0 + kb + 32 + kgb, 0, 0);
        __builtin_prefetch(Bp1 + kb + 32 + kgb, 0, 0);
      }
    }
    v16h af0 = frag_a_vec(Arow0, kb, kg);
    v16h af1 = frag_a_vec(Arow1, kb, kg);
    v16h bf0, bf1;
    if (TRANSB) {
      bf0 = frag_bt_vec(Bp0, kb, kgb);
      bf1 = frag_bt_vec(Bp1, kb, kgb);
    } else {
      bf0 = frag_bn_vec(Bm, ldb, n0 + (lane & 15), kb, kgb);
      bf1 = frag_bn_vec(Bm, ldb, n0 + 16 + (lane & 15), kb, kgb);
    }
    a00 = __builtin_amdgcn_wmma_f32_16x16x32_f16(false, af0, false, bf0, (short)0, a00, false, false);
    a01 = __builtin_amdgcn_wmma_f32_16x16x32_f16(false, af0, false, bf1, (short)0, a01, false, false);
    a10 = __builtin_amdgcn_wmma_f32_16x16x32_f16(false, af1, false, bf0, (short)0, a10, false, false);
    a11 = __builtin_amdgcn_wmma_f32_16x16x32_f16(false, af1, false, bf1, (short)0, a11, false, false);
  }
  int n  = n0 + (lane & 15);
  int mb = m0 + ((lane >> 4) << 3);
  store_tile(a00, C, ldc, mb,      n,      bias, beta, relu);
  store_tile(a01, C, ldc, mb,      n + 16, bias, beta, relu);
  store_tile(a10, C, ldc, mb + 16, n,      bias, beta, relu);
  store_tile(a11, C, ldc, mb + 16, n + 16, bias, beta, relu);
}

// ---------------- guarded generic GEMM (odd shapes: K=954 / N=2) ----------------
// Requires M%16==0 (true for all guarded uses: M=2048); K/N tails handled by
// branch-free clamped loads.
__global__ void gemm_wmma_guarded(const float* __restrict__ A, int lda,
                                  const float* __restrict__ Bm, int ldb, int transB,
                                  const float* __restrict__ bias,
                                  float* __restrict__ C, int ldc,
                                  int M, int N, int K, int beta, int relu) {
  int lane = threadIdx.x & 31;
  int wave = threadIdx.x >> 5;
  int m0 = (blockIdx.y * 4 + wave) * 16;
  int n0 = blockIdx.x * 16;
  if (m0 >= M) return;
  v8f acc = {};
  for (int kb = 0; kb < K; kb += 32) {
    v16h af = frag_a_g(A, lda, m0 + (lane & 15), kb, K, lane);
    v16h bf = frag_b_g(Bm, ldb, n0 + (lane & 15), N, kb, K, lane, transB);
    acc = __builtin_amdgcn_wmma_f32_16x16x32_f16(false, af, false, bf, (short)0, acc, false, false);
  }
  int n = n0 + (lane & 15);
  int mbase = m0 + ((lane >> 4) << 3);
  if (n < N) {
    float bv = bias ? bias[n] : 0.f;
    #pragma unroll
    for (int i = 0; i < 8; ++i) {
      int m = mbase + i;
      float v = acc[i] + bv;
      if (beta) v += C[(size_t)m * ldc + n];
      if (relu && v < 0.f) v = 0.f;
      C[(size_t)m * ldc + n] = v;
    }
  }
}

// ---------------- co-attention S = tanh(lW_b @ r_b^T), pair-masked ----------------
// grid = (4, 4, Bsz), block = 32: one wave per 32x32 tile of the 100x100 scores.
// Loads are unguarded (over-reads land in adjacent workspace, never stored).
__global__ void coattn_s_kernel(const float* __restrict__ lW, const float* __restrict__ r,
                                const float* __restrict__ m1, const float* __restrict__ m2,
                                float* __restrict__ S) {
  int b = blockIdx.z;
  int lane = threadIdx.x & 31;
  int m0 = blockIdx.y * 32, n0 = blockIdx.x * 32;
  const float* Ab = lW + (size_t)b * NPG * HID;
  const float* Rb = r  + (size_t)b * NPG * HID;
  int kg  = (lane >> 4) << 4;
  int kgb = (lane >> 4) << 4;
  const float* Arow0 = Ab + (size_t)(m0 + (lane & 15)) * HID;
  const float* Arow1 = Arow0 + 16 * HID;
  const float* Brow0 = Rb + (size_t)(n0 + (lane & 15)) * HID;
  const float* Brow1 = Brow0 + 16 * HID;
  v8f a00 = {}, a01 = {}, a10 = {}, a11 = {};
  #pragma unroll
  for (int kb = 0; kb < HID; kb += 32) {
    v16h af0 = frag_a_vec(Arow0, kb, kg);
    v16h af1 = frag_a_vec(Arow1, kb, kg);
    v16h bf0 = frag_bt_vec(Brow0, kb, kgb);
    v16h bf1 = frag_bt_vec(Brow1, kb, kgb);
    a00 = __builtin_amdgcn_wmma_f32_16x16x32_f16(false, af0, false, bf0, (short)0, a00, false, false);
    a01 = __builtin_amdgcn_wmma_f32_16x16x32_f16(false, af0, false, bf1, (short)0, a01, false, false);
    a10 = __builtin_amdgcn_wmma_f32_16x16x32_f16(false, af1, false, bf0, (short)0, a10, false, false);
    a11 = __builtin_amdgcn_wmma_f32_16x16x32_f16(false, af1, false, bf1, (short)0, a11, false, false);
  }
  int nA = n0 + (lane & 15);
  int mb = m0 + ((lane >> 4) << 3);
  float* Sb = S + (size_t)b * NPG * NPG;
  const float* m1b = m1 + (size_t)b * NPG;
  const float* m2b = m2 + (size_t)b * NPG;
  #pragma unroll
  for (int tj = 0; tj < 2; ++tj) {
    int n = nA + tj * 16;
    if (n >= NPG) continue;
    float mn = m2b[n];
    #pragma unroll
    for (int ti = 0; ti < 2; ++ti) {
      v8f acc = (tj == 0) ? (ti == 0 ? a00 : a10) : (ti == 0 ? a01 : a11);
      #pragma unroll
      for (int i = 0; i < 8; ++i) {
        int m = mb + ti * 16 + i;
        if (m < NPG) {
          float pair = m1b[m] * mn;
          float v = tanhf(acc[i]);
          Sb[(size_t)m * NPG + n] = (pair > 0.f) ? v : NEGV;
        }
      }
    }
  }
}

// ---------------- elementwise / scatter helpers ----------------
__device__ __forceinline__ void atomicMaxF(float* addr, float v) {
  if (v >= 0.f) atomicMax((int*)addr, __float_as_int(v));
  else          atomicMin((unsigned int*)addr, __float_as_uint(v));
}

__global__ void fill_kernel(float* p, float v, size_t n) {
  size_t i = (size_t)blockIdx.x * blockDim.x + threadIdx.x;
  if (i < n) p[i] = v;
}

__global__ void cell_norm_kernel(const float* __restrict__ cell, float* __restrict__ out) {
  __shared__ float red[256];
  int b = blockIdx.x, t = threadIdx.x;
  const float* cb = cell + (size_t)b * CELLD;
  float s = 0.f;
  for (int i = t; i < CELLD; i += 256) { float v = cb[i]; s += v * v; }
  red[t] = s; __syncthreads();
  for (int st = 128; st > 0; st >>= 1) { if (t < st) red[t] += red[t + st]; __syncthreads(); }
  float inv = 1.f / (sqrtf(red[0]) + 1e-12f);
  float* ob = out + (size_t)b * CELLD;
  for (int i = t; i < CELLD; i += 256) ob[i] = cb[i] * inv;
}

__global__ void add_cell_kernel(const float* __restrict__ x, const float* __restrict__ cexp,
                                float* __restrict__ out) {
  size_t i = (size_t)blockIdx.x * blockDim.x + threadIdx.x;
  if (i >= (size_t)NT * IND) return;
  int col = (int)(i & (IND - 1));
  size_t node = i >> 8;
  int b = (int)(node / NPG);
  out[i] = x[i] + cexp[(size_t)b * IND + col];
}

__global__ void gat_alpha_kernel(const float* __restrict__ hfeat,
                                 const float* __restrict__ a_src, const float* __restrict__ a_dst,
                                 float* __restrict__ as_, float* __restrict__ ad_) {
  int idx = blockIdx.x * blockDim.x + threadIdx.x;
  if (idx >= NT * HEADS) return;
  int node = idx >> 2, h = idx & 3;
  const float* hv = hfeat + (size_t)node * HID + h * DHD;
  float s = 0.f, d = 0.f;
  for (int k = 0; k < DHD; ++k) { s += hv[k] * a_src[h * DHD + k]; d += hv[k] * a_dst[h * DHD + k]; }
  as_[idx] = s; ad_[idx] = d;
}

__device__ __forceinline__ void edge_sd(const int* __restrict__ src, const int* __restrict__ dst,
                                        int e, int& s, int& d) {
  if (e < Ecnt) { s = src[e]; d = dst[e]; } else { s = d = e - Ecnt; }
}

__global__ void gat_edge_max_kernel(const int* __restrict__ src, const int* __restrict__ dst,
                                    const float* __restrict__ as_, const float* __restrict__ ad_,
                                    float* __restrict__ e, float* __restrict__ segmax) {
  int idx = blockIdx.x * blockDim.x + threadIdx.x;
  if (idx >= ETOT * HEADS) return;
  int ei = idx >> 2, h = idx & 3, s, d;
  edge_sd(src, dst, ei, s, d);
  float v = as_[s * HEADS + h] + ad_[d * HEADS + h];
  v = (v > 0.f) ? v : 0.2f * v;                      // leaky_relu 0.2
  e[idx] = v;
  atomicMaxF(&segmax[d * HEADS + h], v);
}

__global__ void gat_edge_exp_kernel(const int* __restrict__ src, const int* __restrict__ dst,
                                    float* __restrict__ e, const float* __restrict__ segmax,
                                    float* __restrict__ segsum) {
  int idx = blockIdx.x * blockDim.x + threadIdx.x;
  if (idx >= ETOT * HEADS) return;
  int ei = idx >> 2, h = idx & 3, s, d;
  edge_sd(src, dst, ei, s, d);
  float ex = __expf(e[idx] - segmax[d * HEADS + h]);
  e[idx] = ex;
  atomicAdd(&segsum[d * HEADS + h], ex);
}

__global__ void gat_edge_aggr_kernel(const int* __restrict__ src, const int* __restrict__ dst,
                                     const float* __restrict__ e, const float* __restrict__ segsum,
                                     const float* __restrict__ hfeat, float* __restrict__ out) {
  int idx = blockIdx.x * blockDim.x + threadIdx.x;
  if (idx >= ETOT * HEADS) return;
  int ei = idx >> 2, h = idx & 3, s, d;
  edge_sd(src, dst, ei, s, d);
  float a = e[idx] / (segsum[d * HEADS + h] + 1e-16f);
  const float* hv = hfeat + (size_t)s * HID + h * DHD;
  float* ov = out + (size_t)d * HID + h * DHD;
  for (int k = 0; k < DHD; ++k) atomicAdd(&ov[k], a * hv[k]);
}

__global__ void deg_count_kernel(const int* __restrict__ dst, float* __restrict__ deg) {
  int idx = blockIdx.x * blockDim.x + threadIdx.x;
  if (idx >= ETOT) return;
  int d = (idx < Ecnt) ? dst[idx] : idx - Ecnt;
  atomicAdd(&deg[d], 1.0f);
}

__global__ void gcn_aggr_kernel(const int* __restrict__ src, const int* __restrict__ dst,
                                const float* __restrict__ deg, const float* __restrict__ xw,
                                float* __restrict__ out) {
  int idx = blockIdx.x * blockDim.x + threadIdx.x;
  if (idx >= ETOT * 4) return;
  int ei = idx >> 2, q = idx & 3, s, d;
  edge_sd(src, dst, ei, s, d);
  float ds_ = deg[s], dd = deg[d];
  float c = ((ds_ > 0.f) ? rsqrtf(ds_) : 0.f) * ((dd > 0.f) ? rsqrtf(dd) : 0.f);
  const float* xv = xw + (size_t)s * HID + q * 32;
  float* ov = out + (size_t)d * HID + q * 32;
  for (int k = 0; k < 32; ++k) atomicAdd(&ov[k], c * xv[k]);
}

__global__ void combine_relu_kernel(const float* __restrict__ g1, const float* __restrict__ b1,
                                    const float* __restrict__ g2, const float* __restrict__ b2,
                                    float* __restrict__ h) {
  size_t i = (size_t)blockIdx.x * blockDim.x + threadIdx.x;
  if (i >= (size_t)NT * HID) return;
  int col = (int)(i & (HID - 1));
  float a = g1[i] + b1[col]; a = a > 0.f ? a : 0.f;
  float c = g2[i] + b2[col]; c = c > 0.f ? c : 0.f;
  h[i] = a + c;
}

__global__ void lstm_point_kernel(const float* __restrict__ g, const float* __restrict__ bias,
                                  float* __restrict__ hstate, float* __restrict__ cstate,
                                  float* __restrict__ rnn, int t) {
  int idx = blockIdx.x * blockDim.x + threadIdx.x;
  if (idx >= Bsz * HID) return;
  int b = idx >> 7, d = idx & 127;
  const float* gb = g + (size_t)b * 4 * HID;
  float i_ = gb[d]           + bias[d];
  float f_ = gb[HID + d]     + bias[HID + d];
  float gg = gb[2 * HID + d] + bias[2 * HID + d];
  float o_ = gb[3 * HID + d] + bias[3 * HID + d];
  float si = 1.f / (1.f + __expf(-i_));
  float sf = 1.f / (1.f + __expf(-f_));
  float so = 1.f / (1.f + __expf(-o_));
  float c = sf * cstate[idx] + si * tanhf(gg);
  float h = so * tanhf(c);
  cstate[idx] = c; hstate[idx] = h;
  rnn[((size_t)b * NPG + t) * HID + d] = h;
}

__global__ void rowcolmax_kernel(const float* __restrict__ S, const float* __restrict__ m1,
                                 const float* __restrict__ m2, float* __restrict__ sl,
                                 float* __restrict__ sr) {
  int b = blockIdx.x, t = threadIdx.x;
  if (t >= NPG) return;
  const float* Sb = S + (size_t)b * NPG * NPG;
  float mx = NEGV;
  for (int j = 0; j < NPG; ++j) mx = fmaxf(mx, Sb[t * NPG + j]);
  sl[b * NPG + t] = (m1[b * NPG + t] > 0.f) ? mx : NEGV;
  float mx2 = NEGV;
  for (int i = 0; i < NPG; ++i) mx2 = fmaxf(mx2, Sb[i * NPG + t]);
  sr[b * NPG + t] = (m2[b * NPG + t] > 0.f) ? mx2 : NEGV;
}

__global__ void softmax100_kernel(float* __restrict__ v) {
  __shared__ float red[128];
  int b = blockIdx.x, t = threadIdx.x;
  float* vb = v + (size_t)b * NPG;
  float x = (t < NPG) ? vb[t] : NEGV;
  red[t] = x; __syncthreads();
  for (int s = 64; s > 0; s >>= 1) { if (t < s) red[t] = fmaxf(red[t], red[t + s]); __syncthreads(); }
  float mx = red[0]; __syncthreads();
  float e = (t < NPG) ? __expf(x - mx) : 0.f;
  red[t] = e; __syncthreads();
  for (int s = 64; s > 0; s >>= 1) { if (t < s) red[t] += red[t + s]; __syncthreads(); }
  float sum = red[0];
  if (t < NPG) vb[t] = e / sum;
}

__global__ void attn_pool_kernel(const float* __restrict__ a, const float* __restrict__ l,
                                 float* __restrict__ out) {
  int b = blockIdx.x, d = threadIdx.x;
  const float* lb = l + (size_t)b * NPG * HID;
  const float* ab = a + (size_t)b * NPG;
  float s = 0.f;
  for (int i = 0; i < NPG; ++i) s += ab[i] * lb[i * HID + d];
  out[(size_t)b * HID + d] = s;
}

__global__ void concat_feat_kernel(const float* pl2, const float* pr2, const float* pl1,
                                   const float* pr1, const float* cellred, float* feat) {
  int idx = blockIdx.x * blockDim.x + threadIdx.x;
  if (idx >= Bsz * FEATD) return;
  int b = idx / FEATD, c = idx % FEATD;
  float v;
  if (c < 128)       v = pl2[b * HID + c];
  else if (c < 256)  v = pr2[b * HID + (c - 128)];
  else if (c < 384)  v = pl1[b * HID + (c - 256)];
  else if (c < 512)  v = pr1[b * HID + (c - 384)];
  else               v = cellred[b * IND + (c - 512)];
  feat[idx] = v;
}

// ---------------- host helpers ----------------
static inline dim3 g1(size_t n, int blk) { return dim3((unsigned)((n + blk - 1) / blk)); }

// fast path: M,N,K multiples of 32, aligned rows
static void gemm_f(hipStream_t st, const float* A, int lda, const float* Bm, int ldb, int transB,
                   const float* bias, float* C, int ldc, int M, int N, int K, int beta, int relu) {
  dim3 grid((N + 127) / 128, (M + 63) / 64);
  if (transB)
    gemm_wmma_fast<1><<<grid, 256, 0, st>>>(A, lda, Bm, ldb, bias, C, ldc, M, N, K, beta, relu);
  else
    gemm_wmma_fast<0><<<grid, 256, 0, st>>>(A, lda, Bm, ldb, bias, C, ldc, M, N, K, beta, relu);
}

static void gemm_g(hipStream_t st, const float* A, int lda, const float* Bm, int ldb, int transB,
                   const float* bias, float* C, int ldc, int M, int N, int K, int beta, int relu) {
  dim3 grid((N + 15) / 16, (M + 63) / 64);
  gemm_wmma_guarded<<<grid, 128, 0, st>>>(A, lda, Bm, ldb, transB, bias, C, ldc, M, N, K, beta, relu);
}

struct Scratch {
  float *tmpA, *tmpB, *gatout, *gcnout, *asrc, *adst, *segmax, *segsum, *deg, *eexp, *gbuf;
};

static void run_layer(hipStream_t st, const float* x, int Kin, const int* ei,
                      const float* gatW, const float* a_src, const float* a_dst, const float* gat_b,
                      const float* gcnW, const float* gcn_b,
                      const float* Whh, const float* Wih, const float* lstm_b,
                      float* h_out, float* rnn, float* hstate, float* cstate, const Scratch& w) {
  const int* src = ei; const int* dst = ei + Ecnt;
  gemm_f(st, x, Kin, gatW, HID, 0, nullptr, w.tmpA, HID, NT, HID, Kin, 0, 0);   // GAT h = x@W
  gemm_f(st, x, Kin, gcnW, HID, 0, nullptr, w.tmpB, HID, NT, HID, Kin, 0, 0);   // GCN xw
  gat_alpha_kernel<<<g1((size_t)NT * HEADS, 256), 256, 0, st>>>(w.tmpA, a_src, a_dst, w.asrc, w.adst);
  fill_kernel<<<g1((size_t)NT * HEADS, 256), 256, 0, st>>>(w.segmax, -3.0e38f, (size_t)NT * HEADS);
  fill_kernel<<<g1((size_t)NT * HEADS, 256), 256, 0, st>>>(w.segsum, 0.f, (size_t)NT * HEADS);
  fill_kernel<<<g1((size_t)NT * HID, 256), 256, 0, st>>>(w.gatout, 0.f, (size_t)NT * HID);
  fill_kernel<<<g1((size_t)NT * HID, 256), 256, 0, st>>>(w.gcnout, 0.f, (size_t)NT * HID);
  fill_kernel<<<g1(NT, 256), 256, 0, st>>>(w.deg, 0.f, NT);
  gat_edge_max_kernel<<<g1((size_t)ETOT * HEADS, 256), 256, 0, st>>>(src, dst, w.asrc, w.adst, w.eexp, w.segmax);
  gat_edge_exp_kernel<<<g1((size_t)ETOT * HEADS, 256), 256, 0, st>>>(src, dst, w.eexp, w.segmax, w.segsum);
  gat_edge_aggr_kernel<<<g1((size_t)ETOT * HEADS, 256), 256, 0, st>>>(src, dst, w.eexp, w.segsum, w.tmpA, w.gatout);
  deg_count_kernel<<<g1(ETOT, 256), 256, 0, st>>>(dst, w.deg);
  gcn_aggr_kernel<<<g1((size_t)ETOT * 4, 256), 256, 0, st>>>(src, dst, w.deg, w.tmpB, w.gcnout);
  combine_relu_kernel<<<g1((size_t)NT * HID, 256), 256, 0, st>>>(w.gatout, gat_b, w.gcnout, gcn_b, h_out);
  // LSTM over T=100 steps: gates = h@Whh^T + x_t@Wih^T + b
  for (int t = 0; t < NPG; ++t) {
    gemm_f(st, hstate, HID, Whh, HID, 1, nullptr, w.gbuf, 4 * HID, Bsz, 4 * HID, HID, 0, 0);
    gemm_f(st, h_out + (size_t)t * HID, NPG * HID, Wih, HID, 1, nullptr, w.gbuf, 4 * HID, Bsz, 4 * HID, HID, 1, 0);
    lstm_point_kernel<<<g1((size_t)Bsz * HID, 256), 256, 0, st>>>(w.gbuf, lstm_b, hstate, cstate, rnn, t);
  }
}

static void run_coattn(hipStream_t st, const float* poolW, const float* l, const float* r,
                       const float* m1, const float* m2, float* pl, float* pr,
                       float* lWbuf, float* Sbuf, float* slb, float* srb) {
  gemm_f(st, l, HID, poolW, HID, 0, nullptr, lWbuf, HID, NT, HID, HID, 0, 0);
  coattn_s_kernel<<<dim3(4, 4, Bsz), 32, 0, st>>>(lWbuf, r, m1, m2, Sbuf);
  rowcolmax_kernel<<<Bsz, 128, 0, st>>>(Sbuf, m1, m2, slb, srb);
  softmax100_kernel<<<Bsz, 128, 0, st>>>(slb);
  softmax100_kernel<<<Bsz, 128, 0, st>>>(srb);
  attn_pool_kernel<<<Bsz, 128, 0, st>>>(slb, l, pl);
  attn_pool_kernel<<<Bsz, 128, 0, st>>>(srb, r, pr);
}

extern "C" void kernel_launch(void* const* d_in, const int* in_sizes, int n_in,
                              void* d_out, int out_size, void* d_ws, size_t ws_size,
                              hipStream_t stream) {
  // inputs: dict order, params flattened with JAX sorted-key pytree order
  const float* x1    = (const float*)d_in[0];
  const int*   ei1   = (const int*)  d_in[1];
  const float* mask1 = (const float*)d_in[3];
  const float* cell  = (const float*)d_in[4];
  const float* x2    = (const float*)d_in[5];
  const int*   ei2   = (const int*)  d_in[6];
  const float* mask2 = (const float*)d_in[8];
  const float* fW1   = (const float*)d_in[9];   // final: W1,W2,b1,b2
  const float* fW2   = (const float*)d_in[10];
  const float* fb1   = (const float*)d_in[11];
  const float* fb2   = (const float*)d_in[12];
  const float* g0W   = (const float*)d_in[13];  // gat[0]: W, a_dst, a_src, b
  const float* g0ad  = (const float*)d_in[14];
  const float* g0as  = (const float*)d_in[15];
  const float* g0b   = (const float*)d_in[16];
  const float* g1Wm  = (const float*)d_in[17];  // gat[1]
  const float* g1ad  = (const float*)d_in[18];
  const float* g1as  = (const float*)d_in[19];
  const float* g1b   = (const float*)d_in[20];
  const float* c0W   = (const float*)d_in[21];  // gcn[0]: W, b
  const float* c0b   = (const float*)d_in[22];
  const float* c1W   = (const float*)d_in[23];
  const float* c1b   = (const float*)d_in[24];
  const float* Whh   = (const float*)d_in[25];  // lstm: Whh, Wih, b
  const float* Wih   = (const float*)d_in[26];
  const float* lb    = (const float*)d_in[27];
  const float* p1W   = (const float*)d_in[28];
  const float* p2W   = (const float*)d_in[29];
  const float* redW  = (const float*)d_in[30];
  const float* redb  = (const float*)d_in[31];
  const float* red2W = (const float*)d_in[32];
  const float* red2b = (const float*)d_in[33];
  float* out = (float*)d_out;

  // ---- carve workspace (all counts multiples of 4 -> 16B-aligned buffers) ----
  float* W = (float*)d_ws;
  size_t off = 0;
  #define CARVE(name, cnt) float* name = W + off; off += (size_t)(cnt);
  CARVE(cell_n,  (size_t)Bsz * CELLD)
  CARVE(cellexp, (size_t)Bsz * IND)
  CARVE(cellred, (size_t)Bsz * IND)
  CARVE(hin_l,   (size_t)NT * IND)
  CARVE(hin_r,   (size_t)NT * IND)
  CARVE(h_l,     (size_t)NT * HID)
  CARVE(h_r,     (size_t)NT * HID)
  CARVE(rnn_l,   (size_t)NT * HID)
  CARVE(rnn_r,   (size_t)NT * HID)
  CARVE(lWbuf,   (size_t)NT * HID)
  CARVE(tmpA,    (size_t)NT * HID)
  CARVE(tmpB,    (size_t)NT * HID)
  CARVE(gatout,  (size_t)NT * HID)
  CARVE(gcnout,  (size_t)NT * HID)
  CARVE(asrc,    (size_t)NT * HEADS)
  CARVE(adst,    (size_t)NT * HEADS)
  CARVE(segmax,  (size_t)NT * HEADS)
  CARVE(segsum,  (size_t)NT * HEADS)
  CARVE(degb,    (size_t)NT)
  CARVE(eexp,    (size_t)ETOT * HEADS)
  CARVE(hst_l,   (size_t)Bsz * HID)
  CARVE(cst_l,   (size_t)Bsz * HID)
  CARVE(hst_r,   (size_t)Bsz * HID)
  CARVE(cst_r,   (size_t)Bsz * HID)
  CARVE(gbuf,    (size_t)Bsz * 4 * HID)
  CARVE(Sbuf,    (size_t)Bsz * NPG * NPG)
  CARVE(slb,     (size_t)Bsz * NPG)
  CARVE(srb,     (size_t)Bsz * NPG)
  CARVE(pl1,     (size_t)Bsz * HID)
  CARVE(pr1,     (size_t)Bsz * HID)
  CARVE(pl2,     (size_t)Bsz * HID)
  CARVE(pr2,     (size_t)Bsz * HID)
  CARVE(featb,   (size_t)Bsz * FEATD)
  CARVE(midb,    (size_t)Bsz * MIDD)
  #undef CARVE
  (void)ws_size; (void)n_in; (void)in_sizes; (void)out_size;

  Scratch w = { tmpA, tmpB, gatout, gcnout, asrc, adst, segmax, segsum, degb, eexp, gbuf };

  // ---- cell pathway (K=954 odd -> guarded GEMM) ----
  cell_norm_kernel<<<Bsz, 256, 0, stream>>>(cell, cell_n);
  gemm_g(stream, cell_n, CELLD, red2W, IND, 0, red2b, cellexp, IND, Bsz, IND, CELLD, 0, 0);
  gemm_g(stream, cell_n, CELLD, redW,  IND, 0, redb,  cellred, IND, Bsz, IND, CELLD, 0, 0);
  add_cell_kernel<<<g1((size_t)NT * IND, 256), 256, 0, stream>>>(x1, cellexp, hin_l);
  add_cell_kernel<<<g1((size_t)NT * IND, 256), 256, 0, stream>>>(x2, cellexp, hin_r);

  // ---- LSTM states ----
  fill_kernel<<<g1((size_t)Bsz * HID, 256), 256, 0, stream>>>(hst_l, 0.f, (size_t)Bsz * HID);
  fill_kernel<<<g1((size_t)Bsz * HID, 256), 256, 0, stream>>>(cst_l, 0.f, (size_t)Bsz * HID);
  fill_kernel<<<g1((size_t)Bsz * HID, 256), 256, 0, stream>>>(hst_r, 0.f, (size_t)Bsz * HID);
  fill_kernel<<<g1((size_t)Bsz * HID, 256), 256, 0, stream>>>(cst_r, 0.f, (size_t)Bsz * HID);

  // ---- two GNN+LSTM layers per side (ping-pong hin <-> h) ----
  run_layer(stream, hin_l, IND, ei1, g0W, g0as, g0ad, g0b, c0W, c0b, Whh, Wih, lb,
            h_l, rnn_l, hst_l, cst_l, w);
  run_layer(stream, h_l,  HID, ei1, g1Wm, g1as, g1ad, g1b, c1W, c1b, Whh, Wih, lb,
            hin_l, rnn_l, hst_l, cst_l, w);
  run_layer(stream, hin_r, IND, ei2, g0W, g0as, g0ad, g0b, c0W, c0b, Whh, Wih, lb,
            h_r, rnn_r, hst_r, cst_r, w);
  run_layer(stream, h_r,  HID, ei2, g1Wm, g1as, g1ad, g1b, c1W, c1b, Whh, Wih, lb,
            hin_r, rnn_r, hst_r, cst_r, w);

  // ---- co-attention pools ----
  run_coattn(stream, p1W, rnn_l, rnn_r, mask1, mask2, pl1, pr1, lWbuf, Sbuf, slb, srb);
  run_coattn(stream, p2W, hin_l, hin_r, mask1, mask2, pl2, pr2, lWbuf, Sbuf, slb, srb);

  // ---- final MLP ----
  concat_feat_kernel<<<g1((size_t)Bsz * FEATD, 256), 256, 0, stream>>>(pl2, pr2, pl1, pr1, cellred, featb);
  gemm_f(stream, featb, FEATD, fW1, MIDD, 0, fb1, midb, MIDD, Bsz, MIDD, FEATD, 0, 1);
  gemm_g(stream, midb, MIDD, fW2, 2, 0, fb2, out, 2, Bsz, 2, MIDD, 0, 0);
}
